// YModel_75118978007346
// MI455X (gfx1250) — matrix operand order repaired and checked by reading, not compiled, not run on hardware
//
#include <hip/hip_runtime.h>

typedef __attribute__((ext_vector_type(16))) _Float16 v16h;
typedef __attribute__((ext_vector_type(8)))  _Float16 v8h;
typedef __attribute__((ext_vector_type(8)))  float    v8f;

#define B_      128
#define H_      80
#define L_      600
#define WSTART  400     // last window start: 20*STRIDE
#define ALPHA   0.2f

__device__ __forceinline__ float leaky(float x) { return x >= 0.f ? x : ALPHA * x; }

// One workgroup per batch image, 256 threads = 8 wave32.
// Phase A: 5x (float4 load of 80x40 slice -> 5x5/s2 VALID conv + leaky -> 2x2 maxpool -> f16 map [8][171])
// Phase B: 5 chain stages; 3x3 SAME conv as 5 tap-pair WMMAs (K = 2 taps x 16 channels) per 16-row M tile.
//          All WMMA operands are contiguous ds_load_b128 with immediate offsets (channel-last layout).
//          Epilogue is unconditional (padded fout absorbs dummy rows / zero columns).
// Phase C: deterministic fixed-order GAP.
__launch_bounds__(256)
__global__ void ymodel_lastwin_fused(
    const float* __restrict__ in,
    const float* __restrict__ spec_w, const float* __restrict__ spec_b,
    const float* __restrict__ w1, const float* __restrict__ b1,
    const float* __restrict__ w2, const float* __restrict__ b2,
    const float* __restrict__ w3, const float* __restrict__ b3,
    const float* __restrict__ w4, const float* __restrict__ b4,
    const float* __restrict__ w5, const float* __restrict__ b5,
    float* __restrict__ out)
{
    __shared__ __align__(16) float    xs[80][40];       // one segment's input slice (f32)
    __shared__ _Float16               spool[5][8][171]; // pooled segment maps (f16)
    __shared__ __align__(16) _Float16 act[21][11][16];  // zero-padded activations, CHANNEL-LAST
    __shared__ __align__(16) _Float16 Wpair[5][32][16]; // per tap-pair: N x K-row weights (flat n*32 + k, N-major)
    __shared__ float                  specw[8][25];
    __shared__ float                  specb[8];
    __shared__ float                  biasv[16];
    __shared__ float                  fout[16][176];    // stage output (f32), padded M and N

    const int b    = blockIdx.x;
    const int tid  = threadIdx.x;
    const int lane = tid & 31;
    const int wave = tid >> 5;
    const int lrow = lane & 15;
    const int hi   = lane >> 4;

    for (int t = tid; t < 200; t += 256) specw[t / 25][t % 25] = spec_w[t];
    if (tid < 8) specb[tid] = spec_b[tid];
    for (int t = tid; t < 21 * 11 * 16; t += 256) (&act[0][0][0])[t] = (_Float16)0.f;
    __syncthreads();

    // ---------------- Phase A: spec conv + leaky + maxpool, 5 segments ----------------
    for (int k = 0; k < 5; ++k) {
        const int c0 = WSTART + 40 * k;
        for (int t = tid; t < 800; t += 256) {        // 80x40 floats as float4
            const int r = t / 10, c4 = (t % 10) * 4;
            *(float4*)&xs[r][c4] = *(const float4*)(in + b * (H_ * L_) + r * L_ + c0 + c4);
        }
        __syncthreads();

        for (int t = tid; t < 8 * 171; t += 256) {
            const int co = t / 171, m = t % 171, ph = m / 9, pw = m % 9;
            float mx = -3.4e38f;
            #pragma unroll
            for (int dy = 0; dy < 2; ++dy)
                #pragma unroll
                for (int dx = 0; dx < 2; ++dx) {
                    const int r0 = 4 * ph + 2 * dy;
                    const int c1 = 4 * pw + 2 * dx;
                    float acc = specb[co];
                    #pragma unroll
                    for (int kh = 0; kh < 5; ++kh)
                        #pragma unroll
                        for (int kw = 0; kw < 5; ++kw)
                            acc += xs[r0 + kh][c1 + kw] * specw[co][kh * 5 + kw];
                    acc = leaky(acc);
                    mx = fmaxf(mx, acc);
                }
            spool[k][co][m] = (_Float16)mx;
        }
        __syncthreads();
    }

    // ---------------- Phase B: 5 chain stages via tap-pair WMMA ----------------
    const float* wj[5] = { w1, w2, w3, w4, w5 };
    const float* bj[5] = { b1, b2, b3, b4, b5 };

    // tap pairs: (0,1)(2,3)(4,5)(6,7)(8,pad) ; tap t -> (ky,kx) = (t/3, t%3)
    constexpr int KY0[5] = { 0, 0, 1, 2, 2 }, KX0[5] = { 0, 2, 1, 0, 2 };
    constexpr int KY1[5] = { 0, 1, 1, 2, 2 }, KX1[5] = { 1, 0, 2, 1, 2 };

    #pragma unroll
    for (int j = 0; j < 5; ++j) {
        const int Cin  = (j == 0) ? 8 : 16;
        const int Cout = (j == 4) ? 4 : 8;

        __syncthreads();  // previous stage's reads of act/Wpair and writes of fout complete

        // Pack weights (N-major): Wpair_flat[p][n*32 + tp*16 + c] = w_j[n][c][tap=2p+tp], zero-padded.
        for (int t = tid; t < 5 * 32 * 16; t += 256) {
            const int p = t >> 9, r = t & 511, kr = r >> 4, n = r & 15;
            const int tp = (p << 1) + (kr >> 4), c = kr & 15;
            _Float16 v = (_Float16)0.f;
            if (tp <= 8 && c < Cin && n < Cout) v = (_Float16)wj[j][(n * Cin + c) * 9 + tp];
            (&Wpair[p][0][0])[n * 32 + kr] = v;
        }
        // Activations (channel-last): ch0-7 = pooled segment j; ch8-15 = previous stage output.
        for (int t = tid; t < 8 * 171; t += 256) {
            const int c = t / 171, m = t % 171, y = m / 9 + 1, x = m % 9 + 1;
            act[y][x][c] = spool[j][c][m];
            if (j > 0) act[y][x][8 + c] = (_Float16)fout[c][m];
        }
        if (tid < 16) biasv[tid] = (tid < Cout) ? bj[j][tid] : 0.f;
        __syncthreads();

        const float bb = biasv[lrow];   // per-lane bias for D column n = lrow (0 for padded cols)

        // 11 M-tiles over M=171; wave-uniform loop => EXEC all ones at every WMMA.
        for (int tile = wave; tile < 11; tile += 8) {
            int m = tile * 16 + lrow;
            if (m > 170) m = 170;            // pad rows duplicate row 170 (stored to pad region)
            const int oy = m / 9, ox = m % 9;

            const _Float16* abase = &act[oy][ox][hi * 8];   // lane base; taps are const offsets
            const _Float16* wrow  = &Wpair[0][0][0] + lrow * 32 + hi * 8;     // lane's N-column base

            v8f acc = { 0.f, 0.f, 0.f, 0.f, 0.f, 0.f, 0.f, 0.f };
            #pragma unroll
            for (int p = 0; p < 5; ++p) {
                v16h af, bf;
                ((v8h*)&af)[0] = *(const v8h*)(abase + (KY0[p] * 11 + KX0[p]) * 16);
                ((v8h*)&af)[1] = *(const v8h*)(abase + (KY1[p] * 11 + KX1[p]) * 16);
                ((v8h*)&bf)[0] = *(const v8h*)(wrow + p * 512);
                ((v8h*)&bf)[1] = *(const v8h*)(wrow + p * 512 + 16);
                acc = __builtin_amdgcn_wmma_f32_16x16x32_f16(
                          false, af, false, bf, (short)0, acc, false, false);
            }

            // D layout: n = lane&15, m = tile*16 + hi*8 + r. Unconditional stores:
            // pad rows (mm>=171) and pad columns (n>=Cout, exactly leaky(0+0)=0) land in fout's
            // padded region / unused channels and are never read back.
            float* frow = &fout[lrow][tile * 16 + hi * 8];
            #pragma unroll
            for (int r = 0; r < 8; ++r)
                frow[r] = leaky(acc[r] + bb);
        }
    }

    // ---------------- Phase C: global average pool (deterministic) ----------------
    __syncthreads();
    if (tid < 4) {
        float s = 0.f;
        for (int m = 0; m < 171; ++m) s += fout[tid][m];
        out[b * 4 + tid] = s * (1.f / 171.f);
    }
}

extern "C" void kernel_launch(void* const* d_in, const int* in_sizes, int n_in,
                              void* d_out, int out_size, void* d_ws, size_t ws_size,
                              hipStream_t stream) {
    (void)in_sizes; (void)n_in; (void)out_size; (void)d_ws; (void)ws_size;
    ymodel_lastwin_fused<<<dim3(B_), dim3(256), 0, stream>>>(
        (const float*)d_in[0],
        (const float*)d_in[1],  (const float*)d_in[2],
        (const float*)d_in[3],  (const float*)d_in[4],
        (const float*)d_in[5],  (const float*)d_in[6],
        (const float*)d_in[7],  (const float*)d_in[8],
        (const float*)d_in[9],  (const float*)d_in[10],
        (const float*)d_in[11], (const float*)d_in[12],
        (float*)d_out);
}
